// MHA_69441031242283
// MI455X (gfx1250) — matrix operand order repaired and checked by reading, not compiled, not run on hardware
//
#include <hip/hip_runtime.h>
#include <math.h>

// Problem constants (reference: B=2, N=1024, D_MODEL=512, NHEAD=8, D_K=64)
#define BB 2
#define NN 1024
#define HH 8
#define DK 64
#define DM 512

typedef __bf16 bf16_t;
typedef __attribute__((ext_vector_type(16))) __bf16        v16bf;
typedef __attribute__((ext_vector_type(8)))  float         v8f;
typedef __attribute__((ext_vector_type(4)))  unsigned int  u32x4;

union FragB16 {
    v16bf v;
    u32x4 q[2];
};

__device__ __forceinline__ v8f wmma_bf16(v16bf a, v16bf b, v8f c) {
    // D = A(16x32 bf16) * B(32x16 bf16) + C(16x16 f32)
    return __builtin_amdgcn_wmma_f32_16x16x32_bf16(
        /*neg_a=*/false, a, /*neg_b=*/false, b,
        /*c_mod=*/(short)0, c, /*reuse_a=*/false, /*reuse_b=*/false);
}

__device__ __forceinline__ v8f v8f_zero() {
    v8f z;
    #pragma unroll
    for (int i = 0; i < 8; ++i) z[i] = 0.0f;
    return z;
}

// Butterfly reductions within a 16-lane half of the wave (C-layout rows live in
// one half), wave32.
__device__ __forceinline__ float half_max(float v) {
    #pragma unroll
    for (int off = 1; off < 16; off <<= 1) v = fmaxf(v, __shfl_xor(v, off, 32));
    return v;
}
__device__ __forceinline__ float half_sum(float v) {
    #pragma unroll
    for (int off = 1; off < 16; off <<= 1) v += __shfl_xor(v, off, 32);
    return v;
}

// ---------------------------------------------------------------------------
// Prep: fp32 -> bf16 conversions
// ---------------------------------------------------------------------------
__global__ void k_f2bf(const float* __restrict__ s, bf16_t* __restrict__ d, int n) {
    int i = blockIdx.x * 256 + threadIdx.x;
    if (i < n) d[i] = (bf16_t)s[i];
}

// q/k/v_proj [H, DM, DK] -> bf16 transposed [H, DK, DM] so WMMA B-fragments are
// pairs of 16-byte loads along contiguous k.
__global__ void k_wT(const float* __restrict__ s, bf16_t* __restrict__ d) {
    int i = blockIdx.x * 256 + threadIdx.x;       // i over H*DK*DM = 262144
    if (i < HH * DK * DM) {
        int h = i >> 15;                          // DK*DM = 32768 = 2^15
        int n = (i >> 9) & (DK - 1);
        int k = i & (DM - 1);
        d[i] = (bf16_t)s[((size_t)h << 15) + (size_t)k * DK + n];
    }
}

// ---------------------------------------------------------------------------
// Per-head projection + LayerNorm.  One wave = 16 rows x 64 cols of one
// (b,h) head.  X:[B,N,DM] bf16, Wt:[H,DK,DM] bf16.
// transposed==0: Out[B,H,N,DK]; transposed==1: Out[B,H,DK,N] (for V).
// ---------------------------------------------------------------------------
__global__ void k_proj_ln(const bf16_t* __restrict__ X,
                          const bf16_t* __restrict__ Wt,
                          const float* __restrict__ lnw,
                          const float* __restrict__ lnb,
                          bf16_t* __restrict__ Out,
                          int transposed) {
    const int lane = threadIdx.x & 31;
    const int half = lane >> 4;
    const int lcol = lane & 15;
    const int w    = blockIdx.x * 4 + (threadIdx.x >> 5);   // 1024 waves total
    const int b    = w / (HH * (NN / 16));
    const int rem  = w % (HH * (NN / 16));
    const int h    = rem / (NN / 16);
    const int rt   = rem % (NN / 16);
    const int row0 = rt * 16;
    const int aoff = half * 8;                              // A-frag K offset

    const bf16_t* xrow = X + ((size_t)b * NN + row0 + lcol) * DM;

    v8f acc[4];
    #pragma unroll
    for (int t = 0; t < 4; ++t) acc[t] = v8f_zero();

    for (int kc = 0; kc < DM / 32; ++kc) {
        FragB16 a;
        const bf16_t* pa = xrow + kc * 32;
        a.q[0] = *(const u32x4*)(pa + aoff);
        a.q[1] = *(const u32x4*)(pa + aoff + 16);
        #pragma unroll
        for (int t = 0; t < 4; ++t) {
            FragB16 bf;
            const bf16_t* pb = Wt + ((size_t)h * DK + t * 16 + lcol) * DM
                                  + kc * 32 + half * 8;
            bf.q[0] = *(const u32x4*)(pb);
            bf.q[1] = *(const u32x4*)(pb + 16);
            acc[t] = wmma_bf16(a.v, bf.v, acc[t]);
        }
    }

    // LayerNorm over the 64-wide rows (row = r + 8*half, cols across 16 lanes
    // x 4 tiles).
    float mu[8], rs[8];
    #pragma unroll
    for (int r = 0; r < 8; ++r) {
        float s  = acc[0][r] + acc[1][r] + acc[2][r] + acc[3][r];
        float s2 = acc[0][r] * acc[0][r] + acc[1][r] * acc[1][r]
                 + acc[2][r] * acc[2][r] + acc[3][r] * acc[3][r];
        s  = half_sum(s);
        s2 = half_sum(s2);
        float m = s * (1.0f / 64.0f);
        float v = s2 * (1.0f / 64.0f) - m * m;
        mu[r] = m;
        rs[r] = rsqrtf(v + 1e-5f);
    }

    #pragma unroll
    for (int t = 0; t < 4; ++t) {
        int col = t * 16 + lcol;
        float lw = lnw[col], lb = lnb[col];
        #pragma unroll
        for (int r = 0; r < 8; ++r) {
            int row = row0 + r + 8 * half;
            float y = (acc[t][r] - mu[r]) * rs[r] * lw + lb;
            size_t o = transposed
                ? ((size_t)(b * HH + h) * DK + col) * NN + row
                : ((size_t)(b * HH + h) * NN + row) * DK + col;
            Out[o] = (bf16_t)y;
        }
    }
}

// ---------------------------------------------------------------------------
// Fused RBF-gated attention (flash-style, online softmax).
// One wave = 16 queries of one (b,h); loops all 1024 keys in 32-key chunks.
// Qh,Kh:[B,H,N,DK] bf16; Vt:[B,H,DK,N] bf16; Z:[B,N,DM] bf16 with the
// (0,2,3,1) interleave f = k*H + h baked in.
// ---------------------------------------------------------------------------
__global__ void k_attn(const bf16_t* __restrict__ Qh,
                       const bf16_t* __restrict__ Kh,
                       const bf16_t* __restrict__ Vt,
                       const float* __restrict__ coords,
                       const unsigned char* __restrict__ padmask,
                       bf16_t* __restrict__ Z) {
    __shared__ bf16_t plds[4][16][32];          // per-wave 16x32 prob tile

    const int wid  = threadIdx.x >> 5;
    const int lane = threadIdx.x & 31;
    const int half = lane >> 4;
    const int lcol = lane & 15;
    const int w    = blockIdx.x * 4 + wid;      // 1024 waves total
    const int b    = w / (HH * (NN / 16));
    const int rem  = w % (HH * (NN / 16));
    const int h    = rem / (NN / 16);
    const int rt   = rem % (NN / 16);
    const int row0 = rt * 16;
    const int aoff = half * 8;

    // Head RBF constants: spread = 3.7 + 16.3*(20^(h/7)-1)/19
    float ls     = exp2f(log2f(20.0f) * (float)h * (1.0f / 7.0f));
    float spread = 3.7f + (20.0f - 3.7f) * (ls - 1.0f) * (1.0f / 19.0f);
    float inv2s2 = 1.0f / (2.0f * spread * spread);
    const float scale = 0.125f;                 // 1/sqrt(64)

    // Preload Q A-fragments (dk 0..31 and 32..63)
    const bf16_t* qrow = Qh + ((size_t)(b * HH + h) * NN + row0 + lcol) * DK;
    FragB16 qa[2];
    #pragma unroll
    for (int d = 0; d < 2; ++d) {
        qa[d].q[0] = *(const u32x4*)(qrow + d * 32 + aoff);
        qa[d].q[1] = *(const u32x4*)(qrow + d * 32 + aoff + 16);
    }

    // Query coords for this lane's C-layout rows (row = r + 8*half)
    float qx[8], qy[8], qz[8];
    #pragma unroll
    for (int r = 0; r < 8; ++r) {
        int row = row0 + r + 8 * half;
        const float* c = coords + ((size_t)b * NN + row) * 3;
        qx[r] = c[0]; qy[r] = c[1]; qz[r] = c[2];
    }

    float mrow[8], lrow[8];
    v8f acc[4];
    #pragma unroll
    for (int r = 0; r < 8; ++r) { mrow[r] = -1e9f; lrow[r] = 0.0f; }
    #pragma unroll
    for (int t = 0; t < 4; ++t) acc[t] = v8f_zero();

    const bf16_t* kbase = Kh + (size_t)(b * HH + h) * NN * DK;
    const bf16_t* vbase = Vt + (size_t)(b * HH + h) * DK * NN;

    for (int kc = 0; kc < NN / 32; ++kc) {
        const int key0 = kc * 32;
        if (key0 + 64 < NN)   // CDNA5 global_prefetch of the K tile two chunks ahead
            __builtin_prefetch(kbase + (size_t)(key0 + 64) * DK, 0, 1);

        // ---- scores: two 16-key tiles, dk accumulated over two WMMAs each
        v8f s[2];
        #pragma unroll
        for (int t = 0; t < 2; ++t) {
            v8f c0 = v8f_zero();
            #pragma unroll
            for (int d = 0; d < 2; ++d) {
                FragB16 kb;
                const bf16_t* pk = kbase + (size_t)(key0 + t * 16 + lcol) * DK
                                         + d * 32 + half * 8;
                kb.q[0] = *(const u32x4*)(pk);
                kb.q[1] = *(const u32x4*)(pk + 16);
                c0 = wmma_bf16(qa[d].v, kb.v, c0);
            }
            s[t] = c0;
        }

        // ---- per-column key coords / padding
        float kx[2], ky[2], kz[2];
        int   pad[2];
        #pragma unroll
        for (int t = 0; t < 2; ++t) {
            int key = key0 + t * 16 + lcol;
            const float* c = coords + ((size_t)b * NN + key) * 3;
            kx[t] = c[0]; ky[t] = c[1]; kz[t] = c[2];
            pad[t] = padmask[(size_t)b * NN + key];
        }

        // ---- RBF gate + band mask + online softmax
        float p0[8], p1[8], fac[8];
        #pragma unroll
        for (int r = 0; r < 8; ++r) {
            float dx0 = qx[r] - kx[0], dy0 = qy[r] - ky[0], dz0 = qz[r] - kz[0];
            float d20 = dx0 * dx0 + dy0 * dy0 + dz0 * dz0;
            float rb0 = __expf(-d20 * inv2s2);
            bool  iv0 = (rb0 < 0.1f) | (rb0 > 0.9f) | (pad[0] != 0);
            float sc0 = iv0 ? -1e9f : s[0][r] * scale * rb0;

            float dx1 = qx[r] - kx[1], dy1 = qy[r] - ky[1], dz1 = qz[r] - kz[1];
            float d21 = dx1 * dx1 + dy1 * dy1 + dz1 * dz1;
            float rb1 = __expf(-d21 * inv2s2);
            bool  iv1 = (rb1 < 0.1f) | (rb1 > 0.9f) | (pad[1] != 0);
            float sc1 = iv1 ? -1e9f : s[1][r] * scale * rb1;

            float tm   = half_max(fmaxf(sc0, sc1));
            float mnew = fmaxf(mrow[r], tm);
            fac[r]     = __expf(mrow[r] - mnew);
            mrow[r]    = mnew;

            p0[r] = (sc0 <= -5e8f) ? 0.0f : __expf(sc0 - mnew);
            p1[r] = (sc1 <= -5e8f) ? 0.0f : __expf(sc1 - mnew);
            float ts = half_sum(p0[r] + p1[r]);
            lrow[r] = lrow[r] * fac[r] + ts;
        }

        // ---- rescale accumulators
        #pragma unroll
        for (int t = 0; t < 4; ++t)
            #pragma unroll
            for (int r = 0; r < 8; ++r) acc[t][r] *= fac[r];

        // ---- repack P: C-layout -> A-layout via LDS
        __syncthreads();
        #pragma unroll
        for (int r = 0; r < 8; ++r) {
            int trow = r + 8 * half;
            plds[wid][trow][lcol]      = (bf16_t)p0[r];
            plds[wid][trow][16 + lcol] = (bf16_t)p1[r];
        }
        __syncthreads();
        FragB16 pa;
        const bf16_t* pl = &plds[wid][lcol][0];
        pa.q[0] = *(const u32x4*)(pl + aoff);
        pa.q[1] = *(const u32x4*)(pl + aoff + 16);

        // ---- A·V: 4 output dim-tiles
        #pragma unroll
        for (int dt = 0; dt < 4; ++dt) {
            FragB16 vb;
            const bf16_t* pv = vbase + (size_t)(dt * 16 + lcol) * NN
                                     + key0 + half * 8;
            vb.q[0] = *(const u32x4*)(pv);
            vb.q[1] = *(const u32x4*)(pv + 16);
            acc[dt] = wmma_bf16(pa.v, vb.v, acc[dt]);
        }
    }

    // ---- finalize: divide by row sum (zero fully-masked rows), write with
    // f = k*H + h interleave
    #pragma unroll
    for (int dt = 0; dt < 4; ++dt) {
        int col = dt * 16 + lcol;
        #pragma unroll
        for (int r = 0; r < 8; ++r) {
            int row = row0 + r + 8 * half;
            float rl = (lrow[r] > 0.0f) ? (1.0f / lrow[r]) : 0.0f;
            float y = acc[dt][r] * rl;
            Z[((size_t)b * NN + row) * DM + col * HH + h] = (bf16_t)y;
        }
    }
}

// ---------------------------------------------------------------------------
// Output projection: Y[row,i] = sum_f Z[row,f] * Wo[i,f] + bias[i]
// One wave = 16 rows x 64 output cols.
// ---------------------------------------------------------------------------
__global__ void k_outproj(const bf16_t* __restrict__ Z,
                          const bf16_t* __restrict__ Wo,
                          const float* __restrict__ bias,
                          float* __restrict__ Y) {
    const int lane = threadIdx.x & 31;
    const int half = lane >> 4;
    const int lcol = lane & 15;
    const int w    = blockIdx.x * 4 + (threadIdx.x >> 5);   // 1024 waves
    const int rt   = w / (DM / 64);                         // 128 row tiles
    const int ib   = w % (DM / 64);                         // 8 col blocks
    const int row0 = rt * 16;
    const int aoff = half * 8;

    const bf16_t* zrow = Z + (size_t)(row0 + lcol) * DM;

    v8f acc[4];
    #pragma unroll
    for (int t = 0; t < 4; ++t) acc[t] = v8f_zero();

    for (int kc = 0; kc < DM / 32; ++kc) {
        FragB16 a;
        const bf16_t* pa = zrow + kc * 32;
        a.q[0] = *(const u32x4*)(pa + aoff);
        a.q[1] = *(const u32x4*)(pa + aoff + 16);
        #pragma unroll
        for (int t = 0; t < 4; ++t) {
            int i = ib * 64 + t * 16 + lcol;
            FragB16 bf;
            const bf16_t* pb = Wo + (size_t)i * DM + kc * 32 + half * 8;
            bf.q[0] = *(const u32x4*)(pb);
            bf.q[1] = *(const u32x4*)(pb + 16);
            acc[t] = wmma_bf16(a.v, bf.v, acc[t]);
        }
    }

    #pragma unroll
    for (int t = 0; t < 4; ++t) {
        int i = ib * 64 + t * 16 + lcol;
        float bi = bias[i];
        #pragma unroll
        for (int r = 0; r < 8; ++r) {
            int row = row0 + r + 8 * half;
            Y[(size_t)row * DM + i] = acc[t][r] + bi;
        }
    }
}

// ---------------------------------------------------------------------------
extern "C" void kernel_launch(void* const* d_in, const int* in_sizes, int n_in,
                              void* d_out, int out_size, void* d_ws, size_t ws_size,
                              hipStream_t stream) {
    (void)in_sizes; (void)n_in; (void)out_size; (void)ws_size;

    const float* q       = (const float*)d_in[0];
    const float* k       = (const float*)d_in[1];
    const float* v       = (const float*)d_in[2];
    const float* coords  = (const float*)d_in[3];
    const unsigned char* mask = (const unsigned char*)d_in[4];   // jnp.bool_
    const float* q_proj  = (const float*)d_in[5];
    const float* k_proj  = (const float*)d_in[6];
    const float* v_proj  = (const float*)d_in[7];
    const float* q_ln_w  = (const float*)d_in[8];
    const float* q_ln_b  = (const float*)d_in[9];
    const float* k_ln_w  = (const float*)d_in[10];
    const float* k_ln_b  = (const float*)d_in[11];
    const float* v_ln_w  = (const float*)d_in[12];
    const float* v_ln_b  = (const float*)d_in[13];
    const float* out_w   = (const float*)d_in[14];
    const float* out_b   = (const float*)d_in[15];
    float* Y = (float*)d_out;

    // Workspace carve-up (all bf16, all sizes 16B-multiple) — 16 MiB total.
    char* ws = (char*)d_ws;
    size_t off = 0;
    auto alloc = [&](size_t elems) {
        bf16_t* p = (bf16_t*)(ws + off);
        off += elems * sizeof(bf16_t);
        return p;
    };
    const size_t nx  = (size_t)BB * NN * DM;     // 1,048,576
    const size_t nw  = (size_t)HH * DK * DM;     //   262,144
    const size_t nh  = (size_t)BB * HH * NN * DK;// 1,048,576
    bf16_t* qb  = alloc(nx);
    bf16_t* kb  = alloc(nx);
    bf16_t* vb  = alloc(nx);
    bf16_t* wqt = alloc(nw);
    bf16_t* wkt = alloc(nw);
    bf16_t* wvt = alloc(nw);
    bf16_t* wob = alloc((size_t)DM * DM);
    bf16_t* Qh  = alloc(nh);
    bf16_t* Kh  = alloc(nh);
    bf16_t* Vt  = alloc(nh);
    bf16_t* Zb  = alloc(nx);

    // 1) conversions
    k_f2bf<<<(int)((nx + 255) / 256), 256, 0, stream>>>(q, qb, (int)nx);
    k_f2bf<<<(int)((nx + 255) / 256), 256, 0, stream>>>(k, kb, (int)nx);
    k_f2bf<<<(int)((nx + 255) / 256), 256, 0, stream>>>(v, vb, (int)nx);
    k_f2bf<<<(DM * DM + 255) / 256, 256, 0, stream>>>(out_w, wob, DM * DM);
    k_wT<<<(int)((nw + 255) / 256), 256, 0, stream>>>(q_proj, wqt);
    k_wT<<<(int)((nw + 255) / 256), 256, 0, stream>>>(k_proj, wkt);
    k_wT<<<(int)((nw + 255) / 256), 256, 0, stream>>>(v_proj, wvt);

    // 2) per-head projection + LayerNorm (1024 waves each, 4 waves/block)
    const int pblocks = (BB * HH * (NN / 16)) / 4;  // 256
    k_proj_ln<<<pblocks, 128, 0, stream>>>(qb, wqt, q_ln_w, q_ln_b, Qh, 0);
    k_proj_ln<<<pblocks, 128, 0, stream>>>(kb, wkt, k_ln_w, k_ln_b, Kh, 0);
    k_proj_ln<<<pblocks, 128, 0, stream>>>(vb, wvt, v_ln_w, v_ln_b, Vt, 1);

    // 3) fused RBF-gated flash attention
    k_attn<<<pblocks, 128, 0, stream>>>(Qh, Kh, Vt, coords, mask, Zb);

    // 4) output projection
    const int oblocks = ((BB * NN / 16) * (DM / 64)) / 4;  // 256
    k_outproj<<<oblocks, 128, 0, stream>>>(Zb, wob, out_b, Y);
}
